// GraphTransformerEncode_33732673143697
// MI455X (gfx1250) — matrix operand order repaired and checked by reading, not compiled, not run on hardware
//
#include <hip/hip_runtime.h>

// ---------------------------------------------------------------------------
// Problem constants (match reference)
// ---------------------------------------------------------------------------
#define NN   50000
#define EE   800000
#define DD   128
#define HH   8
#define DFF  512
#define MPAD 50048            // 391 * 128, >= NN, multiple of 128

typedef __attribute__((ext_vector_type(16))) __bf16 bf16x16;
typedef __attribute__((ext_vector_type(8)))  float  f32x8;

union AFrag { uint4 u[2]; bf16x16 v; };

// ---------------------------------------------------------------------------
// Workspace layout (bytes, all 256-aligned)
// ---------------------------------------------------------------------------
constexpr size_t OFF_WQ   = 0;
constexpr size_t OFF_WK   = OFF_WQ + (size_t)128 * 128 * 2;
constexpr size_t OFF_WV   = OFF_WK + (size_t)128 * 128 * 2;
constexpr size_t OFF_DN   = OFF_WV + (size_t)128 * 128 * 2;
constexpr size_t OFF_F1   = OFF_DN + (size_t)128 * 128 * 2;
constexpr size_t OFF_F2   = OFF_F1 + (size_t)512 * 128 * 2;
constexpr size_t OFF_KS   = OFF_F2 + (size_t)512 * 128 * 2;       // 128 f32
constexpr size_t OFF_NORM = OFF_KS + 512;                          // [MPAD][8] f32
constexpr size_t OFF_XB   = OFF_NORM + (size_t)MPAD * 8 * 4;       // bf16 [MPAD][128]
constexpr size_t OFF_K    = OFF_XB  + (size_t)MPAD * 128 * 2;      // f32  [MPAD][128] (k, later out1)
constexpr size_t OFF_V    = OFF_K   + (size_t)MPAD * 128 * 4;      // f32  (v, later ffn_out)
constexpr size_t OFF_Q    = OFF_V   + (size_t)MPAD * 128 * 4;      // f32  (q, later attn_out)
constexpr size_t OFF_AGG  = OFF_Q   + (size_t)MPAD * 128 * 4;      // f32  agg
constexpr size_t OFF_AGGB = OFF_AGG + (size_t)MPAD * 128 * 4;      // bf16 agg
constexpr size_t OFF_FFNH = OFF_Q;  // bf16 [MPAD][512] reuses (dead) q + agg regions
static_assert((size_t)MPAD * 512 * 2 == 2 * (size_t)MPAD * 128 * 4, "ffnh reuse fits");

// ---------------------------------------------------------------------------
// Helpers
// ---------------------------------------------------------------------------
__device__ __forceinline__ unsigned short f2bf_u(float f) {
    union { float f; unsigned u; } a; a.f = f;
    unsigned r = a.u + 0x7FFFu + ((a.u >> 16) & 1u);   // round-to-nearest-even
    return (unsigned short)(r >> 16);
}

__device__ __forceinline__ float wave_sum32(float v) {
    for (int m = 1; m < 32; m <<= 1) v += __shfl_xor(v, m, 32);
    return v;
}

// ---------------------------------------------------------------------------
// Fill zeros
// ---------------------------------------------------------------------------
__global__ void fill0_kernel(float* p, int n) {
    int i = blockIdx.x * 256 + threadIdx.x;
    if (i < n) p[i] = 0.0f;
}

// ---------------------------------------------------------------------------
// Weight prep: f32 [K][Nout] row-major  ->  bf16 Bt [Nout][K]
// ---------------------------------------------------------------------------
__global__ void prep_w_kernel(const float* __restrict__ w, __bf16* __restrict__ bt,
                              int K, int Nout) {
    int idx = blockIdx.x * 256 + threadIdx.x;
    if (idx >= K * Nout) return;
    int n  = idx / K;
    int kk = idx - n * K;
    unsigned short h = f2bf_u(w[(size_t)kk * Nout + n]);
    bt[idx] = __builtin_bit_cast(__bf16, h);
}

// ---------------------------------------------------------------------------
// LayerNorm1: feature f32 [N][128] -> x bf16 [MPAD][128]  (pad rows zeroed)
// one wave per row, 4 elems/lane
// ---------------------------------------------------------------------------
__global__ __launch_bounds__(256) void ln1_kernel(
        const float* __restrict__ feat, const float* __restrict__ g,
        const float* __restrict__ b, __bf16* __restrict__ xb) {
    int n = blockIdx.x * 8 + (threadIdx.x >> 5);
    if (n >= MPAD) return;
    int lane = threadIdx.x & 31;
    int d0 = lane * 4;
    unsigned short* xr = (unsigned short*)xb + (size_t)n * 128 + d0;
    if (n >= NN) {
        uint2 z; z.x = 0u; z.y = 0u;
        *(uint2*)xr = z;
        return;
    }
    float4 x = *(const float4*)(feat + (size_t)n * 128 + d0);
    float mean = wave_sum32(x.x + x.y + x.z + x.w) * (1.0f / 128.0f);
    float dx0 = x.x - mean, dx1 = x.y - mean, dx2 = x.z - mean, dx3 = x.w - mean;
    float var = wave_sum32(dx0*dx0 + dx1*dx1 + dx2*dx2 + dx3*dx3) * (1.0f / 128.0f);
    float inv = rsqrtf(var + 1e-6f);
    float4 gg = *(const float4*)(g + d0);
    float4 bb = *(const float4*)(b + d0);
    unsigned short h0 = f2bf_u(dx0 * inv * gg.x + bb.x);
    unsigned short h1 = f2bf_u(dx1 * inv * gg.y + bb.y);
    unsigned short h2 = f2bf_u(dx2 * inv * gg.z + bb.z);
    unsigned short h3 = f2bf_u(dx3 * inv * gg.w + bb.w);
    uint2 o; o.x = (unsigned)h0 | ((unsigned)h1 << 16);
    o.y = (unsigned)h2 | ((unsigned)h3 << 16);
    *(uint2*)xr = o;
}

// ---------------------------------------------------------------------------
// Residual + LayerNorm2: out1 = attn_out + feature (f32), h = LN2(out1) bf16
// ---------------------------------------------------------------------------
__global__ __launch_bounds__(256) void resid_ln2_kernel(
        const float* __restrict__ attn, const float* __restrict__ feat,
        const float* __restrict__ g, const float* __restrict__ b,
        float* __restrict__ out1, __bf16* __restrict__ hb) {
    int n = blockIdx.x * 8 + (threadIdx.x >> 5);
    if (n >= MPAD) return;
    int lane = threadIdx.x & 31;
    int d0 = lane * 4;
    unsigned short* hr = (unsigned short*)hb + (size_t)n * 128 + d0;
    if (n >= NN) {
        uint2 z; z.x = 0u; z.y = 0u;
        *(uint2*)hr = z;
        return;
    }
    float4 a = *(const float4*)(attn + (size_t)n * 128 + d0);
    float4 f = *(const float4*)(feat + (size_t)n * 128 + d0);
    float4 o; o.x = a.x + f.x; o.y = a.y + f.y; o.z = a.z + f.z; o.w = a.w + f.w;
    *(float4*)(out1 + (size_t)n * 128 + d0) = o;
    float mean = wave_sum32(o.x + o.y + o.z + o.w) * (1.0f / 128.0f);
    float dx0 = o.x - mean, dx1 = o.y - mean, dx2 = o.z - mean, dx3 = o.w - mean;
    float var = wave_sum32(dx0*dx0 + dx1*dx1 + dx2*dx2 + dx3*dx3) * (1.0f / 128.0f);
    float inv = rsqrtf(var + 1e-6f);
    float4 gg = *(const float4*)(g + d0);
    float4 bb = *(const float4*)(b + d0);
    unsigned short h0 = f2bf_u(dx0 * inv * gg.x + bb.x);
    unsigned short h1 = f2bf_u(dx1 * inv * gg.y + bb.y);
    unsigned short h2 = f2bf_u(dx2 * inv * gg.z + bb.z);
    unsigned short h3 = f2bf_u(dx3 * inv * gg.w + bb.w);
    uint2 ov; ov.x = (unsigned)h0 | ((unsigned)h1 << 16);
    ov.y = (unsigned)h2 | ((unsigned)h3 << 16);
    *(uint2*)hr = ov;
}

// ---------------------------------------------------------------------------
// WMMA GEMM: C[MPAD][NOUT] = A[MPAD][K](bf16) * Bt[NOUT][K](bf16)^T + bias
// block = 256 threads = 8 waves, each wave owns 16 rows; grid = MPAD/128.
// A tile staged in LDS with +16B row pad (bank-conflict-free b128 reads).
// ---------------------------------------------------------------------------
template<int K, int NOUT, bool RELU, bool OUT_BF16>
__global__ __launch_bounds__(256) void gemm_bf16_kernel(
        const __bf16* __restrict__ A, const __bf16* __restrict__ Bt,
        const float* __restrict__ bias, void* __restrict__ Cout) {
    constexpr int KC   = (K < 128) ? K : 128;   // k-chunk staged in LDS
    constexpr int ROWB = KC * 2 + 16;           // padded LDS row stride (bytes)
    constexpr int NT   = NOUT / 16;
    __shared__ __align__(16) unsigned char lds[128 * ROWB];

    const int tid  = threadIdx.x;
    const int wave = tid >> 5;
    const int lane = tid & 31;
    const int m0   = blockIdx.x * 128;
    const int half = (lane < 16) ? 0 : 1;
    const int l15  = lane & 15;
    const int aRow = wave * 16 + l15;   // LDS row this lane reads for A frags
    const int aK0  = half * 16;         // A frag: lanes 0-15 K=0..15, 16-31 K=16..31
    const int bK0  = half * 16;         // B frag: lanes 0-15 K=0..15, 16-31 K=16..31

    auto stage = [&](int kc) {
        const char* gsrc = (const char*)A + ((size_t)m0 * K + (size_t)kc * KC) * 2;
        constexpr int CPR    = KC * 2 / 16;     // 16B chunks per row
        constexpr int CHUNKS = 128 * CPR;
        for (int c = tid; c < CHUNKS; c += 256) {
            int r = c / CPR;
            int o = (c - r * CPR) * 16;
            uint4 d = *(const uint4*)(gsrc + (size_t)r * K * 2 + o);
            *(uint4*)(&lds[r * ROWB + o]) = d;
        }
    };

    auto epilogue = [&](int n0, const f32x8& acc) {
        float bb = bias[n0 + l15];
        int rowbase = m0 + wave * 16 + half * 8;
        int colg = n0 + l15;
        for (int r = 0; r < 8; ++r) {
            float val = acc[r] + bb;
            if (RELU) val = fmaxf(val, 0.0f);
            size_t idx = (size_t)(rowbase + r) * NOUT + colg;
            if constexpr (OUT_BF16)
                ((__bf16*)Cout)[idx] = __builtin_bit_cast(__bf16, f2bf_u(val));
            else
                ((float*)Cout)[idx] = val;
        }
    };

    if constexpr (K <= 128) {
        stage(0);
        __syncthreads();
        for (int nt = 0; nt < NT; ++nt) {
            const int n0 = nt * 16;
            f32x8 acc = {};
            for (int k0 = 0; k0 < K; k0 += 32) {
                AFrag fa, fb;
                const unsigned char* ap = &lds[aRow * ROWB + (k0 + aK0) * 2];
                fa.u[0] = *(const uint4*)(ap);
                fa.u[1] = *(const uint4*)(ap + 16);
                const uint4* bp = (const uint4*)(Bt + (size_t)(n0 + l15) * K + (k0 + bK0));
                fb.u[0] = bp[0];
                fb.u[1] = bp[1];
                acc = __builtin_amdgcn_wmma_f32_16x16x32_bf16(
                        false, fa.v, false, fb.v, (short)0, acc, false, false);
            }
            epilogue(n0, acc);
        }
    } else {
        f32x8 acc[NT];
        for (int nt = 0; nt < NT; ++nt) acc[nt] = {};
        for (int kc = 0; kc < K / KC; ++kc) {
            __syncthreads();
            stage(kc);
            __syncthreads();
            for (int nt = 0; nt < NT; ++nt) {
                const int n0 = nt * 16;
                for (int k0 = 0; k0 < KC; k0 += 32) {
                    AFrag fa, fb;
                    const unsigned char* ap = &lds[aRow * ROWB + (k0 + aK0) * 2];
                    fa.u[0] = *(const uint4*)(ap);
                    fa.u[1] = *(const uint4*)(ap + 16);
                    const uint4* bp = (const uint4*)(Bt + (size_t)(n0 + l15) * K
                                                     + (kc * KC + k0 + bK0));
                    fb.u[0] = bp[0];
                    fb.u[1] = bp[1];
                    acc[nt] = __builtin_amdgcn_wmma_f32_16x16x32_bf16(
                            false, fa.v, false, fb.v, (short)0, acc[nt], false, false);
                }
            }
        }
        for (int nt = 0; nt < NT; ++nt) epilogue(nt * 16, acc[nt]);
    }
}

// ---------------------------------------------------------------------------
// ks_sum[d] = sum_{n<N} k[n][d]
// ---------------------------------------------------------------------------
__global__ void ksum_kernel(const float* __restrict__ k, float* __restrict__ ks) {
    int d = threadIdx.x;                 // 0..127
    float s = 0.0f;
    for (int n = blockIdx.x; n < NN; n += gridDim.x)
        s += k[(size_t)n * 128 + d];
    atomicAdd(&ks[d], s);
}

// ---------------------------------------------------------------------------
// normalizer[n][h] = dot(q[n, h*16:(h+1)*16], ks[h*16:(h+1)*16])
// ---------------------------------------------------------------------------
__global__ void normalizer_kernel(const float* __restrict__ q,
                                  const float* __restrict__ ks,
                                  float* __restrict__ normv) {
    int idx = blockIdx.x * 256 + threadIdx.x;
    if (idx >= NN * HH) return;
    int n = idx >> 3, h = idx & 7;
    const float* qp = q + (size_t)n * 128 + h * 16;
    const float* kp = ks + h * 16;
    float s = 0.0f;
#pragma unroll
    for (int j = 0; j < 16; ++j) s += qp[j] * kp[j];
    normv[idx] = s;
}

// ---------------------------------------------------------------------------
// Edge kernel: one wave per edge (4 dims per lane).
// attn_weight -> d_out slice; msg scatter -> agg (f32 atomics).
// ---------------------------------------------------------------------------
__global__ __launch_bounds__(256) void edge_kernel(
        const int* __restrict__ ei, const int* __restrict__ spv,
        const int* __restrict__ erel,
        const float* __restrict__ q, const float* __restrict__ k,
        const float* __restrict__ v, const float* __restrict__ rel_table,
        const float* __restrict__ spat_table, const float* __restrict__ normv,
        float* __restrict__ aw_out, float* __restrict__ agg) {
    int e = blockIdx.x * 8 + (threadIdx.x >> 5);
    if (e >= EE) return;
    int lane = threadIdx.x & 31;
    int row = ei[e];
    int col = ei[EE + e];
    int rid = erel[e];
    int sid = spv[e];
    int d0 = lane * 4;
    float4 qe = *(const float4*)(q + (size_t)col * 128 + d0);
    float4 kv = *(const float4*)(k + (size_t)row * 128 + d0);
    float4 re = *(const float4*)(rel_table + (size_t)rid * 128 + d0);
    float4 vv = *(const float4*)(v + (size_t)row * 128 + d0);
    float p = (qe.x + re.x) * (kv.x + re.x)
            + (qe.y + re.y) * (kv.y + re.y)
            + (qe.z + re.z) * (kv.z + re.z)
            + (qe.w + re.w) * (kv.w + re.w);
    p += __shfl_xor(p, 1, 32);
    p += __shfl_xor(p, 2, 32);                 // head-sum in every lane of the quad
    int h = lane >> 2;
    const float norm = 0.59460355750136053f;   // 8^-0.25
    float aw = (p * norm + spat_table[(size_t)sid * HH + h])
             / normv[(size_t)col * HH + h];
    if ((lane & 3) == 0) aw_out[(size_t)e * HH + h] = aw;
    float* ap = agg + (size_t)col * 128 + d0;
    atomicAdd(ap + 0, aw * vv.x);
    atomicAdd(ap + 1, aw * vv.y);
    atomicAdd(ap + 2, aw * vv.z);
    atomicAdd(ap + 3, aw * vv.w);
}

// ---------------------------------------------------------------------------
// f32 -> bf16 bulk convert (4 elems/thread)
// ---------------------------------------------------------------------------
__global__ void cvt_bf16_kernel(const float* __restrict__ src,
                                __bf16* __restrict__ dst, int n4) {
    int i = blockIdx.x * 256 + threadIdx.x;
    if (i >= n4) return;
    float4 x = ((const float4*)src)[i];
    uint2 o;
    o.x = (unsigned)f2bf_u(x.x) | ((unsigned)f2bf_u(x.y) << 16);
    o.y = (unsigned)f2bf_u(x.z) | ((unsigned)f2bf_u(x.w) << 16);
    ((uint2*)dst)[i] = o;
}

// ---------------------------------------------------------------------------
// out2 = out1 + ffn_out (final residual) -> d_out
// ---------------------------------------------------------------------------
__global__ void final_kernel(const float* __restrict__ out1,
                             const float* __restrict__ ffo,
                             float* __restrict__ out, int n4) {
    int i = blockIdx.x * 256 + threadIdx.x;
    if (i >= n4) return;
    float4 a = ((const float4*)out1)[i];
    float4 b = ((const float4*)ffo)[i];
    float4 o; o.x = a.x + b.x; o.y = a.y + b.y; o.z = a.z + b.z; o.w = a.w + b.w;
    ((float4*)out)[i] = o;
}

// ---------------------------------------------------------------------------
// Orchestration
// ---------------------------------------------------------------------------
extern "C" void kernel_launch(void* const* d_in, const int* in_sizes, int n_in,
                              void* d_out, int out_size, void* d_ws, size_t ws_size,
                              hipStream_t stream) {
    const float* feature   = (const float*)d_in[0];
    const int*   ei        = (const int*)d_in[1];
    const int*   spv       = (const int*)d_in[2];
    const int*   erel      = (const int*)d_in[3];
    const float* ln1_g     = (const float*)d_in[4];
    const float* ln1_b     = (const float*)d_in[5];
    const float* wq_w      = (const float*)d_in[6];
    const float* wq_b      = (const float*)d_in[7];
    const float* wk_w      = (const float*)d_in[8];
    const float* wk_b      = (const float*)d_in[9];
    const float* wv_w      = (const float*)d_in[10];
    const float* wv_b      = (const float*)d_in[11];
    const float* rel_table = (const float*)d_in[12];
    const float* spat_tab  = (const float*)d_in[13];
    const float* dense_w   = (const float*)d_in[14];
    const float* dense_b   = (const float*)d_in[15];
    const float* ln2_g     = (const float*)d_in[16];
    const float* ln2_b     = (const float*)d_in[17];
    const float* ffn1_w    = (const float*)d_in[18];
    const float* ffn1_b    = (const float*)d_in[19];
    const float* ffn2_w    = (const float*)d_in[20];
    const float* ffn2_b    = (const float*)d_in[21];

    char* ws = (char*)d_ws;
    __bf16* wq_t  = (__bf16*)(ws + OFF_WQ);
    __bf16* wk_t  = (__bf16*)(ws + OFF_WK);
    __bf16* wv_t  = (__bf16*)(ws + OFF_WV);
    __bf16* dn_t  = (__bf16*)(ws + OFF_DN);
    __bf16* f1_t  = (__bf16*)(ws + OFF_F1);
    __bf16* f2_t  = (__bf16*)(ws + OFF_F2);
    float*  ks    = (float*) (ws + OFF_KS);
    float*  normv = (float*) (ws + OFF_NORM);
    __bf16* xb    = (__bf16*)(ws + OFF_XB);
    float*  kf    = (float*) (ws + OFF_K);    // k, later out1
    float*  vf    = (float*) (ws + OFF_V);    // v, later ffn_out
    float*  qf    = (float*) (ws + OFF_Q);    // q, later attn_out
    float*  agg   = (float*) (ws + OFF_AGG);
    __bf16* aggb  = (__bf16*)(ws + OFF_AGGB);
    __bf16* ffnh  = (__bf16*)(ws + OFF_FFNH); // reuses q + agg regions

    float* out    = (float*)d_out;
    float* aw_out = out + (size_t)NN * DD;    // attn_weight slice of d_out

    const dim3 B256(256);
    const int GEMM_GRID = MPAD / 128;         // 391

    // 0) zero agg + ks
    {
        int n = MPAD * 128;
        fill0_kernel<<<(n + 255) / 256, B256, 0, stream>>>(agg, n);
        fill0_kernel<<<1, B256, 0, stream>>>(ks, 128);
    }
    // 1) weight prep (f32 [K][Nout] -> bf16 [Nout][K])
    prep_w_kernel<<<(128 * 128 + 255) / 256, B256, 0, stream>>>(wq_w,    wq_t, 128, 128);
    prep_w_kernel<<<(128 * 128 + 255) / 256, B256, 0, stream>>>(wk_w,    wk_t, 128, 128);
    prep_w_kernel<<<(128 * 128 + 255) / 256, B256, 0, stream>>>(wv_w,    wv_t, 128, 128);
    prep_w_kernel<<<(128 * 128 + 255) / 256, B256, 0, stream>>>(dense_w, dn_t, 128, 128);
    prep_w_kernel<<<(128 * 512 + 255) / 256, B256, 0, stream>>>(ffn1_w,  f1_t, 128, 512);
    prep_w_kernel<<<(512 * 128 + 255) / 256, B256, 0, stream>>>(ffn2_w,  f2_t, 512, 128);
    // 2) LN1 -> x (bf16)
    ln1_kernel<<<MPAD / 8, B256, 0, stream>>>(feature, ln1_g, ln1_b, xb);
    // 3) q, k, v GEMMs (WMMA bf16)
    gemm_bf16_kernel<128, 128, false, false><<<GEMM_GRID, B256, 0, stream>>>(xb, wq_t, wq_b, qf);
    gemm_bf16_kernel<128, 128, false, false><<<GEMM_GRID, B256, 0, stream>>>(xb, wk_t, wk_b, kf);
    gemm_bf16_kernel<128, 128, false, false><<<GEMM_GRID, B256, 0, stream>>>(xb, wv_t, wv_b, vf);
    // 4) ks_sum, normalizer
    ksum_kernel<<<256, dim3(128), 0, stream>>>(kf, ks);
    normalizer_kernel<<<(NN * HH + 255) / 256, B256, 0, stream>>>(qf, ks, normv);
    // 5) per-edge attention + scatter-add aggregation
    edge_kernel<<<EE / 8, B256, 0, stream>>>(ei, spv, erel, qf, kf, vf,
                                             rel_table, spat_tab, normv, aw_out, agg);
    // 6) agg -> bf16, dense GEMM (attn_out overwrites q region)
    cvt_bf16_kernel<<<(MPAD * 128 / 4 + 255) / 256, B256, 0, stream>>>(agg, aggb, MPAD * 128 / 4);
    gemm_bf16_kernel<128, 128, false, false><<<GEMM_GRID, B256, 0, stream>>>(aggb, dn_t, dense_b, qf);
    // 7) residual + LN2 (out1 -> k region, h bf16 -> xb region)
    resid_ln2_kernel<<<MPAD / 8, B256, 0, stream>>>(qf, feature, ln2_g, ln2_b, kf, xb);
    // 8) FFN (ffn1: relu + bf16 out into ffnh; ffn2: f32 out into v region)
    gemm_bf16_kernel<128, 512, true,  true ><<<GEMM_GRID, B256, 0, stream>>>(xb,   f1_t, ffn1_b, ffnh);
    gemm_bf16_kernel<512, 128, false, false><<<GEMM_GRID, B256, 0, stream>>>(ffnh, f2_t, ffn2_b, vf);
    // 9) final residual -> d_out
    final_kernel<<<(NN * 128 / 4 + 255) / 256, B256, 0, stream>>>(kf, vf, out, NN * 128 / 4);
}
